// MultiHeadAttention_76845554860240
// MI455X (gfx1250) — matrix operand order repaired and checked by reading, not compiled, not run on hardware
//
#include <hip/hip_runtime.h>
#include <hip/hip_bf16.h>
#include <stdint.h>

typedef unsigned short u16;
typedef __attribute__((ext_vector_type(16))) __bf16 v16bf;
typedef __attribute__((ext_vector_type(8)))  float  v8f;

static constexpr int B_  = 2;
static constexpr int S_  = 2048;
static constexpr int D_  = 1024;
static constexpr int H_  = 16;
static constexpr int HD_ = 64;

// ---------- small helpers ----------
__device__ __forceinline__ u16 f2bf(float x) {
  uint32_t u = __float_as_uint(x);
  uint32_t r = u + 0x7FFFu + ((u >> 16) & 1u);   // round-to-nearest-even
  return (u16)(r >> 16);
}
__device__ __forceinline__ float bf2f(u16 x) {
  return __uint_as_float(((uint32_t)x) << 16);
}

union Frag16 {
  v16bf v;
  uint4 q[2];
};

// A-matrix 16x32 bf16 fragment (contiguous k order, same as B): lane holds row m = lane%16.
// halves j -> K = kBase + loff + j (j = 0..15)
// with loff = 16 for lanes 16..31.
__device__ __forceinline__ v16bf load_a_frag(const u16* base, int ld, int row,
                                             int kBase, int lane) {
  int m    = row + (lane & 15);
  int loff = (lane & 16) ? 16 : 0;
  const u16* p = base + (size_t)m * ld + kBase + loff;
  Frag16 f;
  f.q[0] = *(const uint4*)(p);
  f.q[1] = *(const uint4*)(p + 8);
  return f.v;
}

// B-matrix 32x16 bf16 fragment: lane holds column n = lane%16.
// lanes 0..15: halves j -> K = kBase + j ; lanes 16..31: K = kBase + 16 + j.
// `base` is stored column-major-for-B: base[n][k] contiguous in k.
__device__ __forceinline__ v16bf load_b_frag(const u16* base, int ld, int col,
                                             int kBase, int lane) {
  int n  = col + (lane & 15);
  int kh = (lane & 16) ? 16 : 0;
  const u16* p = base + (size_t)n * ld + kBase + kh;
  Frag16 f;
  f.q[0] = *(const uint4*)(p);
  f.q[1] = *(const uint4*)(p + 8);
  return f.v;
}

__device__ __forceinline__ v8f wmma_bf16(v16bf a, v16bf b, v8f c) {
  return __builtin_amdgcn_wmma_f32_16x16x32_bf16(
      /*neg_a=*/false, a, /*neg_b=*/false, b,
      /*c_mod=*/(short)0, c, /*reuse_a=*/false, /*reuse_b=*/false);
}

// Async global -> LDS copy, 16 bytes, ASYNCcnt-tracked (CDNA5).
// VDST VGPR carries the wave-relative LDS byte offset (= low 32 bits of the
// generic shared pointer, ISA 10.x).
__device__ __forceinline__ void async_b128(void* lds, const void* gaddr) {
  uint32_t ldsOff = (uint32_t)(uintptr_t)lds;
  asm volatile("global_load_async_to_lds_b128 %0, %1, off"
               :: "v"(ldsOff), "v"(gaddr) : "memory");
}
__device__ __forceinline__ void wait_async0() {
  asm volatile("s_wait_asynccnt 0" ::: "memory");
}

// ---------- preprocessing kernels ----------
__global__ void cvt_bf16(const float* __restrict__ src, u16* __restrict__ dst,
                         size_t n) {
  size_t i = ((size_t)blockIdx.x * blockDim.x + threadIdx.x) * 4;
  if (i + 3 < n) {
    float4 f = *(const float4*)(src + i);
    ushort4 r;
    r.x = f2bf(f.x); r.y = f2bf(f.y); r.z = f2bf(f.z); r.w = f2bf(f.w);
    *(ushort4*)(dst + i) = r;
  }
}

// maskAdd[b][q][k] = (mask==0 ? -1e9 : 0) in bf16: 2x less traffic than the
// int32 mask, which every one of the 16 heads re-reads.
__global__ void cvt_mask(const int* __restrict__ src, u16* __restrict__ dst,
                         size_t n) {
  size_t i = ((size_t)blockIdx.x * blockDim.x + threadIdx.x) * 4;
  if (i + 3 < n) {
    int4 m = *(const int4*)(src + i);
    const u16 neg = f2bf(-1e9f);
    ushort4 r;
    r.x = (m.x == 0) ? neg : (u16)0;
    r.y = (m.y == 0) ? neg : (u16)0;
    r.z = (m.z == 0) ? neg : (u16)0;
    r.w = (m.w == 0) ? neg : (u16)0;
    *(ushort4*)(dst + i) = r;
  }
}

// Wt[n][k] = W[k][n], fp32 -> bf16, 16x16 LDS-tiled so both sides coalesce.
__global__ void transpose_w(const float* __restrict__ src, u16* __restrict__ dst) {
  __shared__ float tile[16][17];
  int n0 = blockIdx.x * 16, k0 = blockIdx.y * 16;
  int tx = threadIdx.x, ty = threadIdx.y;
  tile[ty][tx] = src[(size_t)(k0 + ty) * D_ + n0 + tx];
  __syncthreads();
  dst[(size_t)(n0 + ty) * D_ + k0 + tx] = f2bf(tile[tx][ty]);
}

// biasT[h][q][k] = bias[k][q][h] (fp32 (S,S,H) -> bf16 (H,S,S)).
// Read coalesced over h (tx), write coalesced over k (tx).
__global__ void transpose_bias(const float* __restrict__ src, u16* __restrict__ dst) {
  __shared__ float tile[16][17];
  int q  = blockIdx.y;
  int k0 = blockIdx.x * 16;
  int tx = threadIdx.x, ty = threadIdx.y;
  tile[ty][tx] = src[(size_t)(k0 + ty) * S_ * H_ + (size_t)q * H_ + tx]; // ty=k, tx=h
  __syncthreads();
  // now tx = k_local, ty = h
  dst[((size_t)ty * S_ + q) * S_ + k0 + tx] = f2bf(tile[tx][ty]);
}

// ---------- WMMA GEMM: out[M,N] = X[M,K](bf16) * Wt[N,K](bf16)^T-logical + bias ----------
template <bool OUT_F32>
__global__ __launch_bounds__(256)
void gemm128x64(const u16* __restrict__ X, const u16* __restrict__ Wt,
                const float* __restrict__ bias, void* __restrict__ out,
                int M, int N, int K) {
  const int lane = threadIdx.x & 31;
  const int wid  = threadIdx.x >> 5;          // 8 waves
  const int wm = wid & 3, wn = wid >> 2;      // 4x2 wave grid
  const int m0 = blockIdx.x * 128 + wm * 32;
  const int n0 = blockIdx.y * 64  + wn * 32;

  v8f acc[2][2] = {};
  for (int k = 0; k < K; k += 32) {
    v16bf a0 = load_a_frag(X,  K, m0,      k, lane);
    v16bf a1 = load_a_frag(X,  K, m0 + 16, k, lane);
    v16bf b0 = load_b_frag(Wt, K, n0,      k, lane);
    v16bf b1 = load_b_frag(Wt, K, n0 + 16, k, lane);
    acc[0][0] = wmma_bf16(a0, b0, acc[0][0]);
    acc[0][1] = wmma_bf16(a0, b1, acc[0][1]);
    acc[1][0] = wmma_bf16(a1, b0, acc[1][0]);
    acc[1][1] = wmma_bf16(a1, b1, acc[1][1]);
  }
  const int roff = (lane & 16) ? 8 : 0;
  #pragma unroll
  for (int mi = 0; mi < 2; ++mi)
    #pragma unroll
    for (int ni = 0; ni < 2; ++ni) {
      int n = n0 + ni * 16 + (lane & 15);
      float bv = bias[n];
      #pragma unroll
      for (int v = 0; v < 8; ++v) {
        int m = m0 + mi * 16 + v + roff;
        float val = acc[mi][ni][v] + bv;
        if (OUT_F32) ((float*)out)[(size_t)m * N + n] = val;
        else         ((u16*)out)[(size_t)m * N + n]   = f2bf(val);
      }
    }
}

// ---------- flash attention: one block per (b, h, 64-row q tile) ----------
__global__ __launch_bounds__(256)
void attn_kernel(const u16* __restrict__ Q, const u16* __restrict__ Km,
                 const u16* __restrict__ V, const u16* __restrict__ maskAdd,
                 const u16* __restrict__ biasT, u16* __restrict__ O) {
  __shared__ float Stile[64][64];     // fp32 scores (q x k)
  __shared__ u16   Ptile[64][72];     // bf16 probabilities (q x k), padded
  __shared__ u16   Vt[64][72];        // bf16 V^T (hd x k), padded
  __shared__ u16   biasTile[64][64];  // async-staged bias tile
  __shared__ u16   maskTile[64][64];  // async-staged additive mask tile
  __shared__ float mrow[64], lrow[64], arow[64];

  const int lane = threadIdx.x & 31;
  const int wid  = threadIdx.x >> 5;
  const int wm = wid & 3;             // 16-row q block within tile
  const int wn = wid >> 2;            // 32-col block (k cols / hd cols)
  const int q0 = blockIdx.x * 64;
  const int h  = blockIdx.y;
  const int b  = blockIdx.z;
  const int roff = (lane & 16) ? 8 : 0;

  const size_t headOff = (size_t)h * HD_;
  const u16* Qb  = Q  + ((size_t)b * S_ + q0) * D_ + headOff;
  const u16* Kb0 = Km + (size_t)b * S_ * D_ + headOff;
  const u16* Vb0 = V  + (size_t)b * S_ * D_ + headOff;
  const u16* biasRow = biasT   + ((size_t)h * S_ + q0) * S_;  // + r*S_ + k
  const u16* maskRow = maskAdd + ((size_t)b * S_ + q0) * S_;  // + r*S_ + k

  // Q fragments, kept in VGPRs for the whole k loop (contraction dim = hd).
  v16bf qa[2];
  qa[0] = load_a_frag(Qb, D_, wm * 16, 0,  lane);
  qa[1] = load_a_frag(Qb, D_, wm * 16, 32, lane);

  if (threadIdx.x < 64) { mrow[threadIdx.x] = -1e30f; lrow[threadIdx.x] = 0.f; }
  v8f o[2] = {};
  __syncthreads();

  for (int kt = 0; kt < S_; kt += 64) {
    // ---- async-DMA bias + additive-mask tiles into LDS (ASYNCcnt) ----
    {
      #pragma unroll
      for (int j = 0; j < 2; ++j) {
        int e   = threadIdx.x + j * 256;      // 0..511 -> 512 x 16B = 8KB/tile
        int row = e >> 3, seg = (e & 7) * 8;  // 8 bf16 per b128
        async_b128(&biasTile[row][seg], biasRow + (size_t)row * S_ + kt + seg);
        async_b128(&maskTile[row][seg], maskRow + (size_t)row * S_ + kt + seg);
      }
    }
    // ---- stage V^T tile (hd x kpos) into LDS ----
    {
      int kp  = threadIdx.x >> 2;
      int hd0 = (threadIdx.x & 3) * 16;
      const u16* vp = Vb0 + (size_t)(kt + kp) * D_ + hd0;
      u16 tmp[16];
      *(uint4*)(tmp)     = *(const uint4*)(vp);
      *(uint4*)(tmp + 8) = *(const uint4*)(vp + 8);
      #pragma unroll
      for (int i = 0; i < 16; ++i) Vt[hd0 + i][kp] = tmp[i];
    }
    // ---- scores: S = Q K^T * (1/8) -> LDS ----
    {
      const u16* Kb = Kb0 + (size_t)kt * D_;
      v8f s0 = {}, s1 = {};
      #pragma unroll
      for (int kc = 0; kc < 2; ++kc) {   // hd chunks of 32
        v16bf b0 = load_b_frag(Kb, D_, wn * 32,      kc * 32, lane);
        v16bf b1 = load_b_frag(Kb, D_, wn * 32 + 16, kc * 32, lane);
        s0 = wmma_bf16(qa[kc], b0, s0);
        s1 = wmma_bf16(qa[kc], b1, s1);
      }
      #pragma unroll
      for (int v = 0; v < 8; ++v) {
        int m = wm * 16 + v + roff;
        Stile[m][wn * 32 +      (lane & 15)] = s0[v] * 0.125f;
        Stile[m][wn * 32 + 16 + (lane & 15)] = s1[v] * 0.125f;
      }
    }
    wait_async0();           // bias/mask tiles landed in LDS
    __syncthreads();
    // ---- online softmax: 4 lanes (same wave) per row, 16 cols each ----
    {
      int r  = threadIdx.x >> 2;
      int c0 = (threadIdx.x & 3) * 16;
      float sv[16];
      float mx = -1e30f;
      #pragma unroll
      for (int i = 0; i < 16; ++i) {
        float x = Stile[r][c0 + i] + bf2f(biasTile[r][c0 + i])
                                   + bf2f(maskTile[r][c0 + i]);
        sv[i] = x;
        mx = fmaxf(mx, x);
      }
      mx = fmaxf(mx, __shfl_xor(mx, 1, 4));
      mx = fmaxf(mx, __shfl_xor(mx, 2, 4));
      float mold = mrow[r];
      float mnew = fmaxf(mold, mx);
      float ps = 0.f;
      #pragma unroll
      for (int i = 0; i < 16; ++i) {
        float pp = __expf(sv[i] - mnew);
        ps += pp;
        Ptile[r][c0 + i] = f2bf(pp);
      }
      ps += __shfl_xor(ps, 1, 4);
      ps += __shfl_xor(ps, 2, 4);
      if ((threadIdx.x & 3) == 0) {
        float al = __expf(mold - mnew);
        arow[r] = al;
        lrow[r] = lrow[r] * al + ps;
        mrow[r] = mnew;
      }
    }
    __syncthreads();
    // ---- rescale accumulator, O += P * V ----
    {
      float al[8];
      #pragma unroll
      for (int v = 0; v < 8; ++v) al[v] = arow[wm * 16 + v + roff];
      #pragma unroll
      for (int v = 0; v < 8; ++v) { o[0][v] *= al[v]; o[1][v] *= al[v]; }
      #pragma unroll
      for (int kc = 0; kc < 2; ++kc) {  // kpos chunks of 32
        v16bf pa  = load_a_frag(&Ptile[0][0], 72, wm * 16, kc * 32, lane);
        v16bf vb0 = load_b_frag(&Vt[0][0],    72, wn * 32,      kc * 32, lane);
        v16bf vb1 = load_b_frag(&Vt[0][0],    72, wn * 32 + 16, kc * 32, lane);
        o[0] = wmma_bf16(pa, vb0, o[0]);
        o[1] = wmma_bf16(pa, vb1, o[1]);
      }
    }
    __syncthreads();
  }
  // ---- normalize and store (bf16, layout (B,S,D)) ----
  float li[8];
  #pragma unroll
  for (int v = 0; v < 8; ++v) li[v] = 1.f / lrow[wm * 16 + v + roff];
  #pragma unroll
  for (int t2 = 0; t2 < 2; ++t2) {
    int col = h * HD_ + wn * 32 + t2 * 16 + (lane & 15);
    #pragma unroll
    for (int v = 0; v < 8; ++v) {
      int m = q0 + wm * 16 + v + roff;
      O[((size_t)b * S_ + m) * D_ + col] = f2bf(o[t2][v] * li[v]);
    }
  }
}

// ---------- orchestration ----------
extern "C" void kernel_launch(void* const* d_in, const int* in_sizes, int n_in,
                              void* d_out, int out_size, void* d_ws, size_t ws_size,
                              hipStream_t stream) {
  (void)in_sizes; (void)n_in; (void)out_size; (void)ws_size;
  const float* query = (const float*)d_in[0];
  const float* key_  = (const float*)d_in[1];
  const float* value = (const float*)d_in[2];
  const int*   mask  = (const int*)d_in[3];
  const float* rpb   = (const float*)d_in[4];
  const float* Wq = (const float*)d_in[5];  const float* bq = (const float*)d_in[6];
  const float* Wk = (const float*)d_in[7];  const float* bk = (const float*)d_in[8];
  const float* Wv = (const float*)d_in[9];  const float* bv = (const float*)d_in[10];
  const float* Wo = (const float*)d_in[11]; const float* bo = (const float*)d_in[12];

  const size_t nBSD  = (size_t)B_ * S_ * D_;   // 4,194,304
  const size_t nDD   = (size_t)D_ * D_;        // 1,048,576
  const size_t nBias = (size_t)H_ * S_ * S_;   // 67,108,864
  const size_t nMask = (size_t)B_ * S_ * S_;   // 8,388,608
  // workspace carve-up (~218 MB total, bf16 elements)
  u16* p = (u16*)d_ws;
  u16* Xq  = p; p += nBSD;
  u16* Xk  = p; p += nBSD;
  u16* Xv  = p; p += nBSD;
  u16* Wqt = p; p += nDD;
  u16* Wkt = p; p += nDD;
  u16* Wvt = p; p += nDD;
  u16* Wot = p; p += nDD;
  u16* Qp  = p; p += nBSD;
  u16* Kp  = p; p += nBSD;
  u16* Vp  = p; p += nBSD;
  u16* Ob  = p; p += nBSD;
  u16* maskAdd = p; p += nMask;
  u16* biasT   = p; p += nBias;

  // 1) fp32 -> bf16 activations, additive bf16 mask
  {
    dim3 blk(256), grd((unsigned)(nBSD / 4 / 256));
    cvt_bf16<<<grd, blk, 0, stream>>>(query, Xq, nBSD);
    cvt_bf16<<<grd, blk, 0, stream>>>(key_,  Xk, nBSD);
    cvt_bf16<<<grd, blk, 0, stream>>>(value, Xv, nBSD);
    dim3 grdm((unsigned)(nMask / 4 / 256));
    cvt_mask<<<grdm, blk, 0, stream>>>(mask, maskAdd, nMask);
  }
  // 2) weight transpose+convert: Wt[n][k]
  {
    dim3 blk(16, 16), grd(D_ / 16, D_ / 16);
    transpose_w<<<grd, blk, 0, stream>>>(Wq, Wqt);
    transpose_w<<<grd, blk, 0, stream>>>(Wk, Wkt);
    transpose_w<<<grd, blk, 0, stream>>>(Wv, Wvt);
    transpose_w<<<grd, blk, 0, stream>>>(Wo, Wot);
  }
  // 3) bias transpose: biasT[h][q][k] = bias[k][q][h]
  {
    dim3 blk(16, 16), grd(S_ / 16, S_);
    transpose_bias<<<grd, blk, 0, stream>>>(rpb, biasT);
  }
  // 4) Q/K/V projections (bf16 out)
  {
    dim3 blk(256), grd(B_ * S_ / 128, D_ / 64);
    gemm128x64<false><<<grd, blk, 0, stream>>>(Xq, Wqt, bq, Qp, B_ * S_, D_, D_);
    gemm128x64<false><<<grd, blk, 0, stream>>>(Xk, Wkt, bk, Kp, B_ * S_, D_, D_);
    gemm128x64<false><<<grd, blk, 0, stream>>>(Xv, Wvt, bv, Vp, B_ * S_, D_, D_);
  }
  // 5) fused flash attention
  attn_kernel<<<dim3(S_ / 64, H_, B_), 256, 0, stream>>>(Qp, Kp, Vp, maskAdd,
                                                         biasT, Ob);
  // 6) output projection (fp32 out)
  gemm128x64<true><<<dim3(B_ * S_ / 128, D_ / 64), 256, 0, stream>>>(
      Ob, Wot, bo, d_out, B_ * S_, D_, D_);
}
